// DeSELayer_20538533610074
// MI455X (gfx1250) — matrix operand order repaired and checked by reading, not compiled, not run on hardware
//
#include <hip/hip_runtime.h>
#include <math.h>

// ---- CDNA5 WMMA types ----
typedef __attribute__((ext_vector_type(16))) __bf16 v16bf;
typedef __attribute__((ext_vector_type(8)))  __bf16 v8bf;
typedef __attribute__((ext_vector_type(8)))  float  v8f;

#define CI   512      // channels (mode-0 length I)
#define CJ   14       // H
#define CK   14       // W
#define JK   196      // J*K
#define KP   224      // K padded to 7*32 for wmma bf16 k=32
#define R    4        // CP rank
#define NIT  20
#define CPEPS 1e-6f

__global__ __launch_bounds__(256)
void dese_cp_als_kernel(const float* __restrict__ x,
                        const float* __restrict__ A0,
                        const float* __restrict__ B0,
                        const float* __restrict__ C0,
                        const float* __restrict__ w1,   // [4,4]
                        const float* __restrict__ w2,   // [1,4]
                        const float* __restrict__ fw1,  // [32,512]
                        const float* __restrict__ fw2,  // [512,32]
                        float* __restrict__ out)
{
    // -------- LDS staging (~258 KB, fits in 320 KB/WGP) --------
    __shared__ __align__(64) __bf16 Tl [CI * KP];   // 229376 B  T[i][j*14+k], cols 196..223 zero
    __shared__ __align__(64) __bf16 KRl[KP * 16];   //   7168 B  KR(B,C) padded to [224][16]
    __shared__ float Al [CI * R];                   //   8192 B
    __shared__ float M0l[CI * R];                   //   8192 B  mode-0 MTTKRP result
    __shared__ float Bl [CJ * R], Cl [CK * R];
    __shared__ float AtA[16], BtB[16], CtC[16], Ginv[16];
    __shared__ float M1l[CJ * R], M2l[CK * R];
    __shared__ float sl[CI];                        // rank_fc output
    __shared__ float hl[32];                        // SE hidden
    __shared__ float gl[CI];                        // sigmoid gate
    __shared__ float w1l[16], w2l[4];

    const int tid  = threadIdx.x;       // 256 threads = 8 waves (wave32)
    const int wave = tid >> 5;
    const int lane = tid & 31;
    const int blk  = blockIdx.x;        // sample index

    const float* xb = x + (size_t)blk * (CI * JK);

    // -------- Phase 0: stage T as bf16 in LDS, zero pads, init factors --------
    for (int row = wave; row < CI; row += 8) {
        for (int c0 = lane * 4; c0 < JK; c0 += 128) {           // 196 = 49*4
            float4 v = *reinterpret_cast<const float4*>(xb + row * JK + c0);
            __bf16* dst = &Tl[row * KP + c0];
            dst[0] = (__bf16)v.x; dst[1] = (__bf16)v.y;
            dst[2] = (__bf16)v.z; dst[3] = (__bf16)v.w;
        }
        if (lane < 14) {                                        // zero K-pad cols 196..223
            Tl[row * KP + JK + lane * 2]     = (__bf16)0.f;
            Tl[row * KP + JK + lane * 2 + 1] = (__bf16)0.f;
        }
    }
    // zero whole KR buffer once (rows>=196 and cols 4..15 stay zero forever)
    for (int idx = tid; idx < KP * 16 / 2; idx += 256)
        reinterpret_cast<unsigned int*>(KRl)[idx] = 0u;
    // factor inits (A0 numerically unused by ALS but loaded for fidelity)
    for (int c = tid; c < CI; c += 256) {
        float4 v = *reinterpret_cast<const float4*>(A0 + (size_t)blk * CI * R + c * R);
        Al[c*R+0]=v.x; Al[c*R+1]=v.y; Al[c*R+2]=v.z; Al[c*R+3]=v.w;
    }
    if (tid < CJ) {
        float4 v = *reinterpret_cast<const float4*>(B0 + (size_t)blk * CJ * R + tid * R);
        Bl[tid*R+0]=v.x; Bl[tid*R+1]=v.y; Bl[tid*R+2]=v.z; Bl[tid*R+3]=v.w;
    }
    if (tid >= 16 && tid < 16 + CK) {
        int k = tid - 16;
        float4 v = *reinterpret_cast<const float4*>(C0 + (size_t)blk * CK * R + k * R);
        Cl[k*R+0]=v.x; Cl[k*R+1]=v.y; Cl[k*R+2]=v.z; Cl[k*R+3]=v.w;
    }
    if (tid < 16) w1l[tid] = w1[tid];
    if (tid < 4)  w2l[tid] = w2[tid];
    __syncthreads();

    // initial Grams BtB, CtC
    if (tid < 16) {
        int p = tid >> 2, q = tid & 3; float s = 0.f;
        for (int j = 0; j < CJ; ++j) s += Bl[j*R+p] * Bl[j*R+q];
        BtB[tid] = s;
    } else if (tid < 32) {
        int e = tid - 16, p = e >> 2, q = e & 3; float s = 0.f;
        for (int k = 0; k < CK; ++k) s += Cl[k*R+p] * Cl[k*R+q];
        CtC[e] = s;
    }
    __syncthreads();

    // -------- ALS iterations --------
    for (int it = 0; it < NIT; ++it) {
        // (1) Ginv = inv(BtB .* CtC + eps I)   |  KR(B,C) rebuild (concurrent)
        if (tid == 0) {
            float Mx[4][8];
            for (int p = 0; p < 4; ++p)
                for (int q = 0; q < 4; ++q) {
                    Mx[p][q]   = BtB[p*4+q]*CtC[p*4+q] + (p==q ? CPEPS : 0.f);
                    Mx[p][4+q] = (p==q) ? 1.f : 0.f;
                }
            for (int c = 0; c < 4; ++c) {
                float inv = 1.f / Mx[c][c];
                for (int q = 0; q < 8; ++q) Mx[c][q] *= inv;
                for (int p = 0; p < 4; ++p) if (p != c) {
                    float f = Mx[p][c];
                    for (int q = 0; q < 8; ++q) Mx[p][q] -= f * Mx[c][q];
                }
            }
            for (int p = 0; p < 4; ++p)
                for (int q = 0; q < 4; ++q) Ginv[p*4+q] = Mx[p][4+q];
        }
        if (tid < JK) {
            int j = tid / CK, k = tid % CK;
            #pragma unroll
            for (int r = 0; r < R; ++r)
                KRl[tid*16 + r] = (__bf16)(Bl[j*R+r] * Cl[k*R+r]);
        }
        __syncthreads();

        // (2) mode-0 MTTKRP via v_wmma_f32_16x16x32_bf16:
        //     M0[512,4] = T0[512,196] @ KR[196,4]  (K padded to 224, N padded to 16)
        {
            const int half = lane >> 4;       // 0: K+0, 1: K+8 (A-frag lane-half select)
            const int m    = lane & 15;       // A row within tile / D column index
            for (int t4 = 0; t4 < 4; ++t4) {
                const int row0 = (wave * 4 + t4) * 16;
                v8f acc = {0.f,0.f,0.f,0.f,0.f,0.f,0.f,0.f};
                #pragma unroll
                for (int kc = 0; kc < 7; ++kc) {
                    const int kbase = kc * 32;
                    // A fragment 16x32 bf16 (ISA 7.12.2 layout)
                    const __bf16* arow = &Tl[(row0 + m) * KP + kbase + half * 8];
                    v8bf alo = *reinterpret_cast<const v8bf*>(arow);        // K 0..7 (or 8..15)
                    v8bf ahi = *reinterpret_cast<const v8bf*>(arow + 16);   // K 16..23 (or 24..31)
                    v16bf afrag;
                    #pragma unroll
                    for (int t = 0; t < 8; ++t) { afrag[t] = alo[t]; afrag[t+8] = ahi[t]; }
                    // B fragment 32x16 bf16: lane l holds column N=l%16, K in same order as A fragment
                    v16bf bfrag;
                    #pragma unroll
                    for (int t = 0; t < 8; ++t) {
                        bfrag[t]   = KRl[(kbase + half * 8 + t) * 16 + m];
                        bfrag[t+8] = KRl[(kbase + 16 + half * 8 + t) * 16 + m];
                    }
                    acc = __builtin_amdgcn_wmma_f32_16x16x32_bf16(
                              false, afrag, false, bfrag, (short)0, acc, false, false);
                }
                const int n = lane & 15;
                if (n < R) {                      // only N columns 0..3 are valid
                    const int mbase = row0 + half * 8;
                    #pragma unroll
                    for (int g = 0; g < 8; ++g) M0l[(mbase + g) * R + n] = acc[g];
                }
            }
        }
        __syncthreads();

        // (3) A = M0 @ Ginv   (solve(G, M0^T)^T with symmetric G)
        for (int i = tid; i < CI; i += 256) {
            float m0 = M0l[i*R+0], m1 = M0l[i*R+1], m2 = M0l[i*R+2], m3 = M0l[i*R+3];
            #pragma unroll
            for (int r = 0; r < R; ++r)
                Al[i*R+r] = m0*Ginv[0*4+r] + m1*Ginv[1*4+r] + m2*Ginv[2*4+r] + m3*Ginv[3*4+r];
        }
        __syncthreads();

        // (4) AtA recompute; zero M1
        if (tid < 16) {
            int p = tid >> 2, q = tid & 3; float s = 0.f;
            for (int i = 0; i < CI; ++i) s += Al[i*R+p] * Al[i*R+q];
            AtA[tid] = s;
        }
        if (tid >= 64 && tid < 64 + CJ*R) M1l[tid - 64] = 0.f;
        __syncthreads();

        // (5) mode-1 MTTKRP: M1[j,r] = sum_i A[i,r] * sum_k T[i,j,k]*C[k,r]
        for (int j = 0; j < CJ; ++j) {
            float p0=0.f, p1=0.f, p2=0.f, p3=0.f;
            #pragma unroll
            for (int ii = 0; ii < 2; ++ii) {
                const int i = tid + ii * 256;
                const __bf16* trow = &Tl[i * KP + j * CK];
                float s0=0.f, s1=0.f, s2=0.f, s3=0.f;
                #pragma unroll
                for (int k = 0; k < CK; ++k) {
                    float tv = (float)trow[k];
                    s0 += tv * Cl[k*R+0]; s1 += tv * Cl[k*R+1];
                    s2 += tv * Cl[k*R+2]; s3 += tv * Cl[k*R+3];
                }
                p0 += Al[i*R+0]*s0; p1 += Al[i*R+1]*s1;
                p2 += Al[i*R+2]*s2; p3 += Al[i*R+3]*s3;
            }
            #pragma unroll
            for (int off = 16; off >= 1; off >>= 1) {
                p0 += __shfl_down(p0, off); p1 += __shfl_down(p1, off);
                p2 += __shfl_down(p2, off); p3 += __shfl_down(p3, off);
            }
            if (lane == 0) {
                atomicAdd(&M1l[j*R+0], p0); atomicAdd(&M1l[j*R+1], p1);
                atomicAdd(&M1l[j*R+2], p2); atomicAdd(&M1l[j*R+3], p3);
            }
        }
        __syncthreads();

        // (6) Ginv = inv(AtA .* CtC + eps I)
        if (tid == 0) {
            float Mx[4][8];
            for (int p = 0; p < 4; ++p)
                for (int q = 0; q < 4; ++q) {
                    Mx[p][q]   = AtA[p*4+q]*CtC[p*4+q] + (p==q ? CPEPS : 0.f);
                    Mx[p][4+q] = (p==q) ? 1.f : 0.f;
                }
            for (int c = 0; c < 4; ++c) {
                float inv = 1.f / Mx[c][c];
                for (int q = 0; q < 8; ++q) Mx[c][q] *= inv;
                for (int p = 0; p < 4; ++p) if (p != c) {
                    float f = Mx[p][c];
                    for (int q = 0; q < 8; ++q) Mx[p][q] -= f * Mx[c][q];
                }
            }
            for (int p = 0; p < 4; ++p)
                for (int q = 0; q < 4; ++q) Ginv[p*4+q] = Mx[p][4+q];
        }
        __syncthreads();

        // (7) B = M1 @ Ginv
        if (tid < CJ) {
            float m0=M1l[tid*R+0], m1=M1l[tid*R+1], m2=M1l[tid*R+2], m3=M1l[tid*R+3];
            #pragma unroll
            for (int r = 0; r < R; ++r)
                Bl[tid*R+r] = m0*Ginv[0*4+r] + m1*Ginv[1*4+r] + m2*Ginv[2*4+r] + m3*Ginv[3*4+r];
        }
        __syncthreads();

        // (8) BtB recompute; zero M2
        if (tid < 16) {
            int p = tid >> 2, q = tid & 3; float s = 0.f;
            for (int j = 0; j < CJ; ++j) s += Bl[j*R+p] * Bl[j*R+q];
            BtB[tid] = s;
        }
        if (tid >= 64 && tid < 64 + CK*R) M2l[tid - 64] = 0.f;
        __syncthreads();

        // (9) mode-2 MTTKRP: M2[k,r] = sum_i A[i,r] * sum_j T[i,j,k]*B[j,r]
        for (int k = 0; k < CK; ++k) {
            float p0=0.f, p1=0.f, p2=0.f, p3=0.f;
            #pragma unroll
            for (int ii = 0; ii < 2; ++ii) {
                const int i = tid + ii * 256;
                const __bf16* tcol = &Tl[i * KP + k];
                float s0=0.f, s1=0.f, s2=0.f, s3=0.f;
                #pragma unroll
                for (int j = 0; j < CJ; ++j) {
                    float tv = (float)tcol[j * CK];
                    s0 += tv * Bl[j*R+0]; s1 += tv * Bl[j*R+1];
                    s2 += tv * Bl[j*R+2]; s3 += tv * Bl[j*R+3];
                }
                p0 += Al[i*R+0]*s0; p1 += Al[i*R+1]*s1;
                p2 += Al[i*R+2]*s2; p3 += Al[i*R+3]*s3;
            }
            #pragma unroll
            for (int off = 16; off >= 1; off >>= 1) {
                p0 += __shfl_down(p0, off); p1 += __shfl_down(p1, off);
                p2 += __shfl_down(p2, off); p3 += __shfl_down(p3, off);
            }
            if (lane == 0) {
                atomicAdd(&M2l[k*R+0], p0); atomicAdd(&M2l[k*R+1], p1);
                atomicAdd(&M2l[k*R+2], p2); atomicAdd(&M2l[k*R+3], p3);
            }
        }
        __syncthreads();

        // (10) Ginv = inv(AtA .* BtB + eps I)
        if (tid == 0) {
            float Mx[4][8];
            for (int p = 0; p < 4; ++p)
                for (int q = 0; q < 4; ++q) {
                    Mx[p][q]   = AtA[p*4+q]*BtB[p*4+q] + (p==q ? CPEPS : 0.f);
                    Mx[p][4+q] = (p==q) ? 1.f : 0.f;
                }
            for (int c = 0; c < 4; ++c) {
                float inv = 1.f / Mx[c][c];
                for (int q = 0; q < 8; ++q) Mx[c][q] *= inv;
                for (int p = 0; p < 4; ++p) if (p != c) {
                    float f = Mx[p][c];
                    for (int q = 0; q < 8; ++q) Mx[p][q] -= f * Mx[c][q];
                }
            }
            for (int p = 0; p < 4; ++p)
                for (int q = 0; q < 4; ++q) Ginv[p*4+q] = Mx[p][4+q];
        }
        __syncthreads();

        // (11) C = M2 @ Ginv
        if (tid < CK) {
            float m0=M2l[tid*R+0], m1=M2l[tid*R+1], m2=M2l[tid*R+2], m3=M2l[tid*R+3];
            #pragma unroll
            for (int r = 0; r < R; ++r)
                Cl[tid*R+r] = m0*Ginv[0*4+r] + m1*Ginv[1*4+r] + m2*Ginv[2*4+r] + m3*Ginv[3*4+r];
        }
        __syncthreads();

        // (12) CtC recompute (for next iteration's A step)
        if (tid < 16) {
            int p = tid >> 2, q = tid & 3; float s = 0.f;
            for (int k = 0; k < CK; ++k) s += Cl[k*R+p] * Cl[k*R+q];
            CtC[tid] = s;
        }
        __syncthreads();
    }

    // -------- rank_fc: s[c] = relu(A @ w1^T) @ w2^T --------
    for (int c = tid; c < CI; c += 256) {
        float a0=Al[c*R+0], a1=Al[c*R+1], a2=Al[c*R+2], a3=Al[c*R+3];
        float sc = 0.f;
        #pragma unroll
        for (int s = 0; s < 4; ++s) {
            float y = a0*w1l[s*4+0] + a1*w1l[s*4+1] + a2*w1l[s*4+2] + a3*w1l[s*4+3];
            y = fmaxf(y, 0.f);
            sc += y * w2l[s];
        }
        sl[c] = sc;
    }
    __syncthreads();

    // -------- SE hidden: h[u] = relu(fw1[u,:] . s) , 8 lanes per output u --------
    {
        const int u = tid >> 3;      // 0..31 (lanes 8g..8g+7 of a wave share u)
        const int sub = tid & 7;
        const float* frow = fw1 + u * CI;
        float acc = 0.f;
        for (int c = sub; c < CI; c += 8) acc += frow[c] * sl[c];
        acc += __shfl_down(acc, 4);
        acc += __shfl_down(acc, 2);
        acc += __shfl_down(acc, 1);
        if (sub == 0) hl[u] = fmaxf(acc, 0.f);
    }
    __syncthreads();

    // -------- gate: g[c] = sigmoid(fw2[c,:] . h) --------
    for (int c = tid; c < CI; c += 256) {
        const float* frow = fw2 + c * 32;
        float acc = 0.f;
        #pragma unroll
        for (int u = 0; u < 32; ++u) acc += frow[u] * hl[u];
        gl[c] = 1.f / (1.f + expf(-acc));
    }
    __syncthreads();

    // -------- out = x * g[c]  (x re-read from L2, f32-exact) --------
    float* ob = out + (size_t)blk * (CI * JK);
    for (int row = wave; row < CI; row += 8) {
        const float gv = gl[row];
        for (int c0 = lane * 4; c0 < JK; c0 += 128) {
            float4 v = *reinterpret_cast<const float4*>(xb + row * JK + c0);
            float4 o; o.x = v.x * gv; o.y = v.y * gv; o.z = v.z * gv; o.w = v.w * gv;
            *reinterpret_cast<float4*>(ob + row * JK + c0) = o;
        }
    }
}

extern "C" void kernel_launch(void* const* d_in, const int* in_sizes, int n_in,
                              void* d_out, int out_size, void* d_ws, size_t ws_size,
                              hipStream_t stream) {
    (void)in_sizes; (void)n_in; (void)d_ws; (void)ws_size; (void)out_size;
    const float* x   = (const float*)d_in[0];
    const float* A0  = (const float*)d_in[1];
    const float* B0  = (const float*)d_in[2];
    const float* C0  = (const float*)d_in[3];
    const float* w1  = (const float*)d_in[4];
    const float* w2  = (const float*)d_in[5];
    const float* fw1 = (const float*)d_in[6];
    const float* fw2 = (const float*)d_in[7];
    float* out = (float*)d_out;
    dese_cp_als_kernel<<<256, 256, 0, stream>>>(x, A0, B0, C0, w1, w2, fw1, fw2, out);
}
